// WordPhraseGraphV1_12979391168959
// MI455X (gfx1250) — compile-verified, matrix operand order checked by reading, not run on hardware
//
#include <hip/hip_runtime.h>
#include <hip/hip_bf16.h>

// ---------------------------------------------------------------------------
// Problem constants (match reference)
// ---------------------------------------------------------------------------
#define P_CNT 4096
#define E_CNT 65536
#define HDIM  512
#define W_CNT 8192
#define EPS_F 1e-6f
#define INV_SQRT_H 0.044194173824159216f   // 1/sqrt(512)

typedef __attribute__((ext_vector_type(16))) __bf16 v16bf;
typedef __attribute__((ext_vector_type(8)))  float  v8f;

// round-to-nearest-even f32 -> bf16 (as raw ushort)
__device__ __forceinline__ unsigned short f2bf(float f) {
    unsigned int u = __float_as_uint(f);
    u = (u + 0x7FFFu + ((u >> 16) & 1u)) >> 16;
    return (unsigned short)u;
}

// ---------------------------------------------------------------------------
// WMMA fragment loaders (wave32 layouts per CDNA5 ISA 7.12.2)
// A (16x32 bf16): lane -> row m = lane&15; half i -> k = 16*(i>>3) + 8*(lane>>4) + (i&7)
// B is pre-swizzled to the mirrored layout so each lane reads 32 contiguous bytes.
// ---------------------------------------------------------------------------
__device__ __forceinline__ v16bf load_afrag(const unsigned short* base, int stride,
                                            int kbase, int lane) {
    const int m  = lane & 15;
    const int kh = (lane & 16) ? 8 : 0;
    const unsigned short* row = base + m * stride + kbase + kh;
    union { unsigned int w[8]; v16bf v; } u;
#pragma unroll
    for (int j = 0; j < 8; ++j) {
        const int k = ((j & 4) << 2) + ((j & 3) << 1);  // pairs: {0,2,4,6,16,18,20,22}
        u.w[j] = *(const unsigned int*)(row + k);
    }
    return u.v;
}

__device__ __forceinline__ v16bf load_bfrag(const unsigned short* p) {
    union { uint4 q[2]; v16bf v; } u;
    const uint4* q = (const uint4*)p;
    u.q[0] = q[0];
    u.q[1] = q[1];
    return u.v;
}

__device__ __forceinline__ v8f wmma_bf16(v16bf a, v16bf b, v8f c) {
    return __builtin_amdgcn_wmma_f32_16x16x32_bf16(false, a, false, b, (short)0, c,
                                                   false, false);
}

// ---------------------------------------------------------------------------
// Weight pre-swizzle: W[K,512] f32 (row-major) -> bf16 fragments
// frag index t = ((kt*32 + nt)*32 + lane)*16 + i
// element    = W[(kt*32 + kmap(i,lane)) * 512 + nt*16 + (lane&15)]
// ---------------------------------------------------------------------------
__global__ __launch_bounds__(256) void k_swizzle_w(const float* __restrict__ W,
                                                   unsigned short* __restrict__ dst,
                                                   int total) {
    const int t = blockIdx.x * 256 + threadIdx.x;
    if (t >= total) return;
    const int i    = t & 15;
    const int lane = (t >> 4) & 31;
    const int nt   = (t >> 9) & 31;
    const int kt   = t >> 14;
    const int k = kt * 32 + ((i & 8) << 1) + ((lane & 16) ? 8 : 0) + (i & 7);
    const int n = nt * 16 + (lane & 15);
    dst[t] = f2bf(W[(size_t)k * HDIM + n]);
}

// ---------------------------------------------------------------------------
// GEMM1: updated_rel[E,512] = [phr_s | phr_o | rel] @ W_rel + b_rel
// block = 16 edge rows, 4 waves, each wave owns 8 N-tiles (128 cols)
// ---------------------------------------------------------------------------
__global__ __launch_bounds__(128) void k_rel_gemm(
    const float* __restrict__ phrase, const float* __restrict__ rel,
    const int* __restrict__ sub, const int* __restrict__ obj,
    const unsigned short* __restrict__ WrelF, const float* __restrict__ b_rel,
    float* __restrict__ out_rel) {
    __shared__ unsigned short As[16 * 1536];
    __shared__ int subs[16], objs[16];
    const int tid = threadIdx.x;
    const int e0  = blockIdx.x * 16;
    if (tid < 16) { subs[tid] = sub[e0 + tid]; objs[tid] = obj[e0 + tid]; }
    __syncthreads();
    for (int idx = tid; idx < 16 * 1536; idx += 128) {
        const int r = idx / 1536;
        const int c = idx - r * 1536;
        float v;
        if (c < 512)       v = phrase[(size_t)subs[r] * HDIM + c];
        else if (c < 1024) v = phrase[(size_t)objs[r] * HDIM + (c - 512)];
        else               v = rel[(size_t)(e0 + r) * HDIM + (c - 1024)];
        As[idx] = f2bf(v);
    }
    __syncthreads();

    const int lane = tid & 31;
    const int wave = tid >> 5;
    v8f acc[8] = {};
    for (int kt = 0; kt < 48; ++kt) {
        const v16bf a = load_afrag(As, 1536, kt * 32, lane);
#pragma unroll
        for (int j = 0; j < 8; ++j) {
            const int nt = wave * 8 + j;
            const v16bf b =
                load_bfrag(WrelF + ((size_t)(kt * 32 + nt) * 32 + lane) * 16);
            acc[j] = wmma_bf16(a, b, acc[j]);
        }
    }
    const int hi = (lane >> 4) & 1;
    const int nc = lane & 15;
#pragma unroll
    for (int j = 0; j < 8; ++j) {
        const int col = (wave * 8 + j) * 16 + nc;
        const float bb = b_rel[col];
#pragma unroll
        for (int r = 0; r < 8; ++r) {
            const int row = e0 + r + hi * 8;
            out_rel[(size_t)row * HDIM + col] = acc[j][r] + bb;
        }
    }
}

// ---------------------------------------------------------------------------
// Fused GEMM2+GEMM3+dot:
// atte[e] = sum_n (([phr_s|ur]@W_sub + b_sub) * ([phr_o|ur]@W_obj + b_obj)) / sqrt(H)
// Never materializes trans_sub / trans_obj.
// ---------------------------------------------------------------------------
__global__ __launch_bounds__(128) void k_att_dot(
    const float* __restrict__ phrase, const float* __restrict__ out_rel,
    const int* __restrict__ sub, const int* __restrict__ obj,
    const unsigned short* __restrict__ WsubF, const float* __restrict__ b_sub,
    const unsigned short* __restrict__ WobjF, const float* __restrict__ b_obj,
    float* __restrict__ atte) {
    __shared__ unsigned short Ps[16 * 512];
    __shared__ unsigned short Po[16 * 512];
    __shared__ unsigned short Ur[16 * 512];
    __shared__ int subs[16], objs[16];
    __shared__ float dsum[4][16];
    const int tid = threadIdx.x;
    const int e0  = blockIdx.x * 16;
    if (tid < 16) { subs[tid] = sub[e0 + tid]; objs[tid] = obj[e0 + tid]; }
    __syncthreads();
    for (int idx = tid; idx < 16 * 512; idx += 128) {
        const int r = idx >> 9;
        const int c = idx & 511;
        Ps[idx] = f2bf(phrase[(size_t)subs[r] * HDIM + c]);
        Po[idx] = f2bf(phrase[(size_t)objs[r] * HDIM + c]);
        Ur[idx] = f2bf(out_rel[(size_t)(e0 + r) * HDIM + c]);
    }
    __syncthreads();

    const int lane = tid & 31;
    const int wave = tid >> 5;
    v8f dot = {};
#pragma unroll 1
    for (int j = 0; j < 8; ++j) {
        const int nt = wave * 8 + j;
        v8f accs = {};
        v8f acco = {};
        for (int kt = 0; kt < 32; ++kt) {
            v16bf as, ao;
            if (kt < 16) {
                as = load_afrag(Ps, 512, kt * 32, lane);
                ao = load_afrag(Po, 512, kt * 32, lane);
            } else {
                as = load_afrag(Ur, 512, (kt - 16) * 32, lane);
                ao = as;
            }
            const v16bf bs =
                load_bfrag(WsubF + ((size_t)(kt * 32 + nt) * 32 + lane) * 16);
            const v16bf bo =
                load_bfrag(WobjF + ((size_t)(kt * 32 + nt) * 32 + lane) * 16);
            accs = wmma_bf16(as, bs, accs);
            acco = wmma_bf16(ao, bo, acco);
        }
        const int col = nt * 16 + (lane & 15);
        const float bsv = b_sub[col];
        const float bov = b_obj[col];
#pragma unroll
        for (int r = 0; r < 8; ++r) dot[r] += (accs[r] + bsv) * (acco[r] + bov);
    }
    // reduce over the 16 lanes of each half (N dimension)
#pragma unroll
    for (int r = 0; r < 8; ++r) {
        float v = dot[r];
        v += __shfl_xor(v, 1);
        v += __shfl_xor(v, 2);
        v += __shfl_xor(v, 4);
        v += __shfl_xor(v, 8);
        dot[r] = v;
    }
    if ((lane & 15) == 0) {
        const int hi = (lane >> 4) & 1;
#pragma unroll
        for (int r = 0; r < 8; ++r) dsum[wave][hi * 8 + r] = dot[r];
    }
    __syncthreads();
    if (tid < 16) {
        const float s = dsum[0][tid] + dsum[1][tid] + dsum[2][tid] + dsum[3][tid];
        atte[e0 + tid] = s * INV_SQRT_H;
    }
}

// ---------------------------------------------------------------------------
// Softmax stats. rowmax/colmax are zero-initialized; the reference max runs over
// the whole (mostly zero) row, so max >= 0 -> int-bit atomicMax is exact for v>0.
// ---------------------------------------------------------------------------
__global__ __launch_bounds__(256) void k_stats1(
    const float* __restrict__ atte, const int* __restrict__ sub,
    const int* __restrict__ obj, float* __restrict__ rowmax,
    float* __restrict__ colmax, int* __restrict__ involved) {
    const int e = blockIdx.x * 256 + threadIdx.x;
    if (e >= E_CNT) return;
    const float v = atte[e];
    const int s = sub[e];
    const int o = obj[e];
    if (v > 0.f) {
        atomicMax((int*)(rowmax + s), __float_as_int(v));
        atomicMax((int*)(colmax + o), __float_as_int(v));
    }
    involved[s] = 1;
    involved[o] = 1;
}

__global__ __launch_bounds__(256) void k_stats2(
    const float* __restrict__ atte, const int* __restrict__ sub,
    const int* __restrict__ obj, const float* __restrict__ rowmax,
    const float* __restrict__ colmax, float* __restrict__ rowsum,
    float* __restrict__ colsum, float* __restrict__ wsub_un,
    float* __restrict__ wobj_un) {
    const int e = blockIdx.x * 256 + threadIdx.x;
    if (e >= E_CNT) return;
    const float v  = atte[e];
    const int s = sub[e];
    const int o = obj[e];
    const float ws = expf(v - rowmax[s]);
    const float wo = expf(v - colmax[o]);
    atomicAdd(rowsum + s, ws);
    atomicAdd(colsum + o, wo);
    wsub_un[e] = ws;
    wobj_un[e] = wo;
}

// ---------------------------------------------------------------------------
// Sparse context accumulation (segment sums over edges):
//   ctxA[s] += w_s * phrase[o];  ctxA[o] += w_o * phrase[s]
//   ctxB[s] += w_s * ur[e];      ctxB[o] += w_o * ur[e]
// ---------------------------------------------------------------------------
__global__ __launch_bounds__(128) void k_ctx(
    const float* __restrict__ phrase, const float* __restrict__ out_rel,
    const int* __restrict__ sub, const int* __restrict__ obj,
    const float* __restrict__ rowsum, const float* __restrict__ colsum,
    const float* __restrict__ wsub_un, const float* __restrict__ wobj_un,
    float* __restrict__ ctxA, float* __restrict__ ctxB) {
    const int e = blockIdx.x;
    const int s = sub[e];
    const int o = obj[e];
    const float wsn = wsub_un[e] / (rowsum[s] + EPS_F);
    const float won = wobj_un[e] / (colsum[o] + EPS_F);
    const float* ur = out_rel + (size_t)e * HDIM;
    for (int c = threadIdx.x; c < HDIM; c += 128) {
        atomicAdd(ctxA + (size_t)s * HDIM + c, wsn * phrase[(size_t)o * HDIM + c]);
        atomicAdd(ctxA + (size_t)o * HDIM + c, won * phrase[(size_t)s * HDIM + c]);
        const float u = ur[c];
        atomicAdd(ctxB + (size_t)s * HDIM + c, wsn * u);
        atomicAdd(ctxB + (size_t)o * HDIM + c, won * u);
    }
}

// ---------------------------------------------------------------------------
// GEMM4 + epilogue: out = involved ? phrase + [ctxA|ctxB]@W_trans + b_trans : phrase
// ---------------------------------------------------------------------------
__global__ __launch_bounds__(128) void k_trans_gemm(
    const float* __restrict__ phrase, const float* __restrict__ ctxA,
    const float* __restrict__ ctxB, const unsigned short* __restrict__ WtF,
    const float* __restrict__ b_trans, const int* __restrict__ involved,
    float* __restrict__ out_phr) {
    __shared__ unsigned short Aq[16 * 1024];
    const int tid = threadIdx.x;
    const int p0  = blockIdx.x * 16;
    for (int idx = tid; idx < 16 * 1024; idx += 128) {
        const int r = idx >> 10;
        const int c = idx & 1023;
        const float v = (c < 512) ? ctxA[(size_t)(p0 + r) * HDIM + c]
                                  : ctxB[(size_t)(p0 + r) * HDIM + (c - 512)];
        Aq[idx] = f2bf(v);
    }
    __syncthreads();

    const int lane = tid & 31;
    const int wave = tid >> 5;
    v8f acc[8] = {};
    for (int kt = 0; kt < 32; ++kt) {
        const v16bf a = load_afrag(Aq, 1024, kt * 32, lane);
#pragma unroll
        for (int j = 0; j < 8; ++j) {
            const int nt = wave * 8 + j;
            const v16bf b =
                load_bfrag(WtF + ((size_t)(kt * 32 + nt) * 32 + lane) * 16);
            acc[j] = wmma_bf16(a, b, acc[j]);
        }
    }
    const int hi = (lane >> 4) & 1;
    const int nc = lane & 15;
#pragma unroll
    for (int j = 0; j < 8; ++j) {
        const int col = (wave * 8 + j) * 16 + nc;
        const float bt = b_trans[col];
#pragma unroll
        for (int r = 0; r < 8; ++r) {
            const int row = p0 + r + hi * 8;
            const float base = phrase[(size_t)row * HDIM + col];
            out_phr[(size_t)row * HDIM + col] =
                involved[row] ? (base + acc[j][r] + bt) : base;
        }
    }
}

// ---------------------------------------------------------------------------
// Host-side launcher
// ---------------------------------------------------------------------------
extern "C" void kernel_launch(void* const* d_in, const int* in_sizes, int n_in,
                              void* d_out, int out_size, void* d_ws, size_t ws_size,
                              hipStream_t stream) {
    (void)in_sizes; (void)n_in; (void)out_size; (void)ws_size;
    const float* word    = (const float*)d_in[0];
    const float* phrase  = (const float*)d_in[1];
    const float* rel     = (const float*)d_in[2];
    const int*   conn    = (const int*)d_in[3];
    const float* W_rel   = (const float*)d_in[4];
    const float* b_rel   = (const float*)d_in[5];
    const float* W_sub   = (const float*)d_in[6];
    const float* b_sub   = (const float*)d_in[7];
    const float* W_obj   = (const float*)d_in[8];
    const float* b_obj   = (const float*)d_in[9];
    const float* W_trans = (const float*)d_in[10];
    const float* b_trans = (const float*)d_in[11];
    const int* sub = conn;
    const int* obj = conn + E_CNT;

    float* out      = (float*)d_out;
    float* out_word = out;
    float* out_phr  = out + (size_t)W_CNT * HDIM;
    float* out_rel  = out_phr + (size_t)P_CNT * HDIM;

    // workspace carve-up (all 256B aligned by construction)
    char* ws = (char*)d_ws;
    unsigned short* WrelF = (unsigned short*)ws;          ws += (size_t)1536 * 512 * 2;
    unsigned short* WsubF = (unsigned short*)ws;          ws += (size_t)1024 * 512 * 2;
    unsigned short* WobjF = (unsigned short*)ws;          ws += (size_t)1024 * 512 * 2;
    unsigned short* WtF   = (unsigned short*)ws;          ws += (size_t)1024 * 512 * 2;
    float* atte    = (float*)ws;                          ws += (size_t)E_CNT * 4;
    float* rowmax  = (float*)ws;                          ws += (size_t)P_CNT * 4;
    float* colmax  = (float*)ws;                          ws += (size_t)P_CNT * 4;
    float* rowsum  = (float*)ws;                          ws += (size_t)P_CNT * 4;
    float* colsum  = (float*)ws;                          ws += (size_t)P_CNT * 4;
    float* wsub_un = (float*)ws;                          ws += (size_t)E_CNT * 4;
    float* wobj_un = (float*)ws;                          ws += (size_t)E_CNT * 4;
    float* ctxA    = (float*)ws;                          ws += (size_t)P_CNT * HDIM * 4;
    float* ctxB    = (float*)ws;                          ws += (size_t)P_CNT * HDIM * 4;
    int*   involved = (int*)ws;                           ws += (size_t)P_CNT * 4;

    // zero accumulators (rowmax..colsum contiguous; ctxA+ctxB contiguous)
    hipMemsetAsync(rowmax, 0, (size_t)4 * P_CNT * 4, stream);
    hipMemsetAsync(ctxA, 0, (size_t)2 * P_CNT * HDIM * 4, stream);
    hipMemsetAsync(involved, 0, (size_t)P_CNT * 4, stream);

    // pass-through word_feat
    hipMemcpyAsync(out_word, word, (size_t)W_CNT * HDIM * 4,
                   hipMemcpyDeviceToDevice, stream);

    // pre-swizzle weights into WMMA B-fragment layout (bf16)
    k_swizzle_w<<<(1536 * 512) / 256, 256, 0, stream>>>(W_rel, WrelF, 1536 * 512);
    k_swizzle_w<<<(1024 * 512) / 256, 256, 0, stream>>>(W_sub, WsubF, 1024 * 512);
    k_swizzle_w<<<(1024 * 512) / 256, 256, 0, stream>>>(W_obj, WobjF, 1024 * 512);
    k_swizzle_w<<<(1024 * 512) / 256, 256, 0, stream>>>(W_trans, WtF, 1024 * 512);

    // GEMM1 -> updated_rel (third output)
    k_rel_gemm<<<E_CNT / 16, 128, 0, stream>>>(phrase, rel, sub, obj, WrelF, b_rel,
                                               out_rel);
    // fused GEMM2+GEMM3+dot -> atte_e
    k_att_dot<<<E_CNT / 16, 128, 0, stream>>>(phrase, out_rel, sub, obj, WsubF,
                                              b_sub, WobjF, b_obj, atte);
    // sparse masked softmax stats
    k_stats1<<<E_CNT / 256, 256, 0, stream>>>(atte, sub, obj, rowmax, colmax,
                                              involved);
    k_stats2<<<E_CNT / 256, 256, 0, stream>>>(atte, sub, obj, rowmax, colmax,
                                              rowsum, colsum, wsub_un, wobj_un);
    // context segment sums
    k_ctx<<<E_CNT, 128, 0, stream>>>(phrase, out_rel, sub, obj, rowsum, colsum,
                                     wsub_un, wobj_un, ctxA, ctxB);
    // GEMM4 + residual + involved mask -> phrase output
    k_trans_gemm<<<P_CNT / 16, 128, 0, stream>>>(phrase, ctxA, ctxB, WtF, b_trans,
                                                 involved, out_phr);
}
